// CorrRatio_18631568130521
// MI455X (gfx1250) — matrix operand order repaired and checked by reading, not compiled, not run on hardware
//
#include <hip/hip_runtime.h>

typedef __attribute__((ext_vector_type(16))) _Float16 v16h;
typedef __attribute__((ext_vector_type(8)))  float    v8f;

#define NBINS   32
#define PRETERM 961.0f          // (NBINS-1)^2
#define EPSV    1e-5f
#define NVOX    (96*96*96)      // 884736, divisible by 32

// workspace layout (floats):
// [0..31]  T1  (dir1: bins on y_pred, intensity y_true)
// [32..63] S1
// [64..95] T2  (dir2: bins on y_true, intensity y_pred)
// [96..127]S2
// [128] sum(y_true) [129] sum(y_true^2) [130] sum(y_pred) [131] sum(y_pred^2)
#define WS_FLOATS 132

__device__ __forceinline__ void atomAddF(float* p, float v) {
    __hip_atomic_fetch_add(p, v, __ATOMIC_RELAXED, __HIP_MEMORY_SCOPE_AGENT);
}

__global__ void cr_zero_ws(float* ws) {
    int i = threadIdx.x;
    if (i < WS_FLOATS) ws[i] = 0.0f;
}

__global__ __launch_bounds__(256) void cr_main(const float* __restrict__ y_true,
                                               const float* __restrict__ y_pred,
                                               float* __restrict__ ws) {
    const int lane   = threadIdx.x & 31;
    const int gwid   = blockIdx.x * (blockDim.x >> 5) + (threadIdx.x >> 5);
    const int nWaves = gridDim.x * (blockDim.x >> 5);
    const int nChunks = NVOX / 32;

    // WMMA f32 accumulators: row0 = T partials, row1 = S partials (cols = bins)
    v8f c1lo = {}, c1hi = {}, c2lo = {}, c2hi = {};
    float st = 0.f, stt = 0.f, sp = 0.f, spp = 0.f;

    const int   binId   = lane & 15;            // B column = bin (per half)
    const int   rowM    = lane & 15;            // A row id
    const int   rbase   = lane & 16;            // unused; B half holds voxels koff..koff+7, koff+16..koff+23
    const int   koff    = (lane & 16) >> 1;     // A K-offset: 0 or 8
    const float binc_lo = (float)binId        * (1.0f / 31.0f);
    const float binc_hi = (float)(binId + 16) * (1.0f / 31.0f);

    for (int ch = gwid; ch < nChunks; ch += nWaves) {
        const int base = ch * 32;
        const float tv = y_true[base + lane];
        const float pv = y_pred[base + lane];
        st += tv; stt += tv * tv;
        sp += pv; spp += pv * pv;

        // ---- B fragments: Parzen weights, lane = bin column -------------
        v16h bT_lo, bT_hi, bP_lo, bP_hi;
#pragma unroll
        for (int j = 0; j < 16; ++j) {
            const float yp = __shfl(pv, (j < 8) ? (koff + j) : (koff + 8 + j), 32);   // voxel koff+j (j<8) / koff+8+j (j>=8)
            const float yt = __shfl(tv, (j < 8) ? (koff + j) : (koff + 8 + j), 32);
            float d;
            d = yp - binc_lo; bP_lo[j] = (_Float16)__expf(-PRETERM * d * d);
            d = yp - binc_hi; bP_hi[j] = (_Float16)__expf(-PRETERM * d * d);
            d = yt - binc_lo; bT_lo[j] = (_Float16)__expf(-PRETERM * d * d);
            d = yt - binc_hi; bT_hi[j] = (_Float16)__expf(-PRETERM * d * d);
        }

        // ---- A fragments: row0 = x, row1 = 1, rows 2..15 = 0 ------------
        v16h aT, aP;
#pragma unroll
        for (int e = 0; e < 8; ++e) {
            const float xt0 = __shfl(tv, koff + e, 32);        // K = koff+e
            const float xt1 = __shfl(tv, koff + 16 + e, 32);   // K = koff+16+e
            const float xp0 = __shfl(pv, koff + e, 32);
            const float xp1 = __shfl(pv, koff + 16 + e, 32);
            const _Float16 one = (_Float16)1.0f, zero = (_Float16)0.0f;
            aT[e]     = (rowM == 0) ? (_Float16)xt0 : ((rowM == 1) ? one : zero);
            aT[e + 8] = (rowM == 0) ? (_Float16)xt1 : ((rowM == 1) ? one : zero);
            aP[e]     = (rowM == 0) ? (_Float16)xp0 : ((rowM == 1) ? one : zero);
            aP[e + 8] = (rowM == 0) ? (_Float16)xp1 : ((rowM == 1) ? one : zero);
        }

        // ---- D = A x B + C (f32 accumulate in registers) ----------------
        // dir1: X = y_true, bins on y_pred
        c1lo = __builtin_amdgcn_wmma_f32_16x16x32_f16(false, aT, false, bP_lo,
                                                      (short)0, c1lo, false, false);
        c1hi = __builtin_amdgcn_wmma_f32_16x16x32_f16(false, aT, false, bP_hi,
                                                      (short)0, c1hi, false, false);
        // dir2: X = y_pred, bins on y_true
        c2lo = __builtin_amdgcn_wmma_f32_16x16x32_f16(false, aP, false, bT_lo,
                                                      (short)0, c2lo, false, false);
        c2hi = __builtin_amdgcn_wmma_f32_16x16x32_f16(false, aP, false, bT_hi,
                                                      (short)0, c2hi, false, false);
    }

    // ---- drain accumulators: row0 (c[0]) = T, row1 (c[1]) = S, lanes 0..15 = bins
    if (lane < 16) {
        atomAddF(&ws[ 0 + lane],      c1lo[0]);
        atomAddF(&ws[32 + lane],      c1lo[1]);
        atomAddF(&ws[ 0 + 16 + lane], c1hi[0]);
        atomAddF(&ws[32 + 16 + lane], c1hi[1]);
        atomAddF(&ws[64 + lane],      c2lo[0]);
        atomAddF(&ws[96 + lane],      c2lo[1]);
        atomAddF(&ws[64 + 16 + lane], c2hi[0]);
        atomAddF(&ws[96 + 16 + lane], c2hi[1]);
    }

    // ---- scalar moment sums: wave reduction then one atomic per wave ----
#pragma unroll
    for (int off = 16; off >= 1; off >>= 1) {
        st  += __shfl_down(st,  off, 32);
        stt += __shfl_down(stt, off, 32);
        sp  += __shfl_down(sp,  off, 32);
        spp += __shfl_down(spp, off, 32);
    }
    if (lane == 0) {
        atomAddF(&ws[128], st);
        atomAddF(&ws[129], stt);
        atomAddF(&ws[130], sp);
        atomAddF(&ws[131], spp);
    }
}

__global__ void cr_finalize(const float* __restrict__ ws, float* __restrict__ out) {
    if (threadIdx.x != 0 || blockIdx.x != 0) return;
    const float Nf = (float)NVOX;
    const float sum_t = ws[128], ss_t = ws[129];
    const float sum_p = ws[130], ss_p = ws[131];
    const float mean_t = sum_t / Nf;
    const float var_t  = (ss_t - Nf * mean_t * mean_t) / (Nf - 1.0f);  // ddof=1
    const float mean_p = sum_p / Nf;
    const float var_p  = (ss_p - Nf * mean_p * mean_p) / (Nf - 1.0f);

    float bg1 = 0.f, tot1 = 0.f, bg2 = 0.f, tot2 = 0.f;
    for (int k = 0; k < NBINS; ++k) {
        // dir1
        float S = ws[32 + k], T = ws[k];
        float m = T / (S + EPSV);
        float d = m - mean_t;
        bg1 += S * d * d; tot1 += S;
        // dir2
        S = ws[96 + k]; T = ws[64 + k];
        m = T / (S + EPSV);
        d = m - mean_p;
        bg2 += S * d * d; tot2 += S;
    }
    const float eta1 = (bg1 / (tot1 + EPSV)) / (var_t + EPSV);
    const float eta2 = (bg2 / (tot2 + EPSV)) / (var_p + EPSV);
    const float cr = eta1 / 3.0f + eta2 / 3.0f;   // CR(yt,yp) + CR(yp,yt)
    out[0] = -cr * 0.5f;
}

extern "C" void kernel_launch(void* const* d_in, const int* in_sizes, int n_in,
                              void* d_out, int out_size, void* d_ws, size_t ws_size,
                              hipStream_t stream) {
    const float* y_true = (const float*)d_in[0];
    const float* y_pred = (const float*)d_in[1];
    float* ws  = (float*)d_ws;
    float* out = (float*)d_out;

    cr_zero_ws<<<1, 256, 0, stream>>>(ws);
    cr_main<<<512, 256, 0, stream>>>(y_true, y_pred, ws);
    cr_finalize<<<1, 32, 0, stream>>>(ws, out);
}
